// MultiHeadAttention_5317169513040
// MI455X (gfx1250) — matrix operand order repaired and checked by reading, not compiled, not run on hardware
//
#include <hip/hip_runtime.h>
#include <hip/hip_bf16.h>

typedef __attribute__((ext_vector_type(16))) _Float16 v16h;
typedef __attribute__((ext_vector_type(2)))  _Float16 h2;
typedef __attribute__((ext_vector_type(8)))  float    v8f;

#define SQ    2048
#define KVLEN 2080   // 2048 real + 16 persistent + 16 zero-pad (masked)
#define HEADS 8
#define HD    64
#define EMB   512

__device__ __forceinline__ v8f wmma_f16(v16h a, v16h b, v8f c) {
  return __builtin_amdgcn_wmma_f32_16x16x32_f16(
      /*neg_a=*/false, a, /*neg_b=*/false, b,
      /*c_mod=*/(short)0, c, /*reuse_a=*/false, /*reuse_b=*/false);
}

// A fragment (16x32 f16, row-major source, row = caller-supplied, K pairs per
// ISA layout: vgpr p<4 -> K=2p+8*half ; p>=4 -> K=16+2(p-4)+8*half)
__device__ __forceinline__ v16h load_frag_a(const _Float16* src, int ld,
                                            int row, int k0, int h8) {
  v16h a;
#pragma unroll
  for (int p = 0; p < 8; ++p) {
    int kb = (p < 4) ? (2 * p + h8) : (16 + 2 * (p - 4) + h8);
    h2 pr = *reinterpret_cast<const h2*>(src + (size_t)row * ld + k0 + kb);
    a[2 * p]     = pr.x;
    a[2 * p + 1] = pr.y;
  }
  return a;
}

// B fragment (32x16 f16). Source given as B^T row-major: element (k, n) at
// bt[n*ld + k]; contraction pairs contiguous. Lane layout matches A fragment:
// vgpr p<4 -> K=2p+8*half ; p>=4 -> K=16+2(p-4)+8*half.
__device__ __forceinline__ v16h load_frag_bt(const _Float16* bt, int ld,
                                             int nrow, int k0, int h16) {
  v16h b;
#pragma unroll
  for (int p = 0; p < 8; ++p) {
    int kb = (p < 4) ? (2 * p + (h16 >> 1)) : (16 + 2 * (p - 4) + (h16 >> 1));
    h2 pr = *reinterpret_cast<const h2*>(bt + (size_t)nrow * ld + k0 + kb);
    b[2 * p]     = pr.x;
    b[2 * p + 1] = pr.y;
  }
  return b;
}

// ---------------------------------------------------------------------------
// Kernel 1: per-head-dim projections q/k/v  (memory-bound, VALU fp32),
// stores f16 in WMMA-friendly layouts:
//   qh [n][h][s][d], kh [n][h][key][d], vT [n][h][d][key]
// ---------------------------------------------------------------------------
__global__ __launch_bounds__(256) void proj_kernel(
    const float* __restrict__ vals, const float* __restrict__ keys,
    const float* __restrict__ qrs,  const float* __restrict__ Wv,
    const float* __restrict__ Wk,   const float* __restrict__ Wq,
    _Float16* __restrict__ qh, _Float16* __restrict__ kh,
    _Float16* __restrict__ vT) {
  __shared__ float sWq[64 * 64], sWk[64 * 64], sWv[64 * 64];
  __shared__ float sq[EMB], sk[EMB], sv[EMB];
  const int tid = threadIdx.x;
  const int tok = blockIdx.x;           // 0 .. N*S-1
  const int n = tok >> 11, s = tok & 2047;
  for (int i = tid; i < 4096; i += 256) {
    sWq[i] = Wq[i]; sWk[i] = Wk[i]; sWv[i] = Wv[i];
  }
  const size_t base = (size_t)tok * EMB;
  for (int i = tid; i < EMB; i += 256) {
    sq[i] = qrs[base + i]; sk[i] = keys[base + i]; sv[i] = vals[base + i];
  }
  __syncthreads();
  for (int o = tid; o < EMB; o += 256) {
    const int h = o >> 6, e = o & 63;
    float aq = 0.f, ak = 0.f, av = 0.f;
#pragma unroll 8
    for (int d = 0; d < 64; ++d) {
      aq += sq[h * 64 + d] * sWq[e * 64 + d];
      ak += sk[h * 64 + d] * sWk[e * 64 + d];
      av += sv[h * 64 + d] * sWv[e * 64 + d];
    }
    qh[((size_t)(n * HEADS + h) * SQ + s) * HD + e]       = (_Float16)aq;
    kh[((size_t)(n * HEADS + h) * KVLEN + s) * HD + e]    = (_Float16)ak;
    vT[((size_t)(n * HEADS + h) * HD + e) * KVLEN + s]    = (_Float16)av;
  }
}

// Kernel 1b: append persistent tokens (keys 2048..2063) + zero pad (2064..2079)
__global__ void persist_kernel(const float* __restrict__ pk,
                               const float* __restrict__ pv,
                               _Float16* __restrict__ kh,
                               _Float16* __restrict__ vT) {
  int idx = blockIdx.x * 256 + threadIdx.x;  // n(3b) h(3b) slot(5b) d(6b)
  if (idx >= 8 * 8 * 32 * 64) return;
  const int d = idx & 63;
  const int slot = (idx >> 6) & 31;
  const int h = (idx >> 11) & 7;
  const int n = idx >> 14;
  float kk = 0.f, vv = 0.f;
  if (slot < 16) {                       // pk/pv: [P][H][D]
    kk = pk[(slot * HEADS + h) * HD + d];
    vv = pv[(slot * HEADS + h) * HD + d];
  }
  kh[((size_t)(n * HEADS + h) * KVLEN + SQ + slot) * HD + d] = (_Float16)kk;
  vT[((size_t)(n * HEADS + h) * HD + d) * KVLEN + SQ + slot] = (_Float16)vv;
}

__global__ void cvt_wo_kernel(const float* __restrict__ Wo,
                              _Float16* __restrict__ WoH) {
  int i = blockIdx.x * 256 + threadIdx.x;
  if (i < EMB * EMB) WoH[i] = (_Float16)Wo[i];
}

// ---------------------------------------------------------------------------
// Kernel 2: fused flash attention with talking heads + ALiBi + persistent KV.
// Grid: N * (S/16) blocks, 256 threads (8 waves). Wave w == head w.
// ---------------------------------------------------------------------------
__global__ __launch_bounds__(256) void attn_kernel(
    const _Float16* __restrict__ qh, const _Float16* __restrict__ kh,
    const _Float16* __restrict__ vT, const int* __restrict__ mask,
    const float* __restrict__ pre_th, const float* __restrict__ post_th,
    _Float16* __restrict__ xh) {
  __shared__ float    ldsE[HEADS][16][32];   // raw per-head energies (16 KB)
  __shared__ _Float16 ldsP[HEADS][16][32];   // probs for A-layout reload (8 KB)
  __shared__ float    ldsO[HEADS][16][64];   // normalized outputs (32 KB)

  const int tid  = threadIdx.x;
  const int w    = tid >> 5;         // wave == source/dest head
  const int lane = tid & 31;
  const int hlf  = lane >> 4;
  const int l16  = lane & 15;
  const int h8   = hlf * 8;
  const int h16  = hlf * 16;

  const int n  = blockIdx.x >> 7;    // /128
  const int q0 = (blockIdx.x & 127) * 16;

  const _Float16* qhW = qh + (size_t)(n * HEADS + w) * SQ * HD;
  const _Float16* khW = kh + (size_t)(n * HEADS + w) * KVLEN * HD;
  const _Float16* vTW = vT + (size_t)(n * HEADS + w) * HD * KVLEN;
  const int* maskN = mask + (size_t)n * SQ * SQ;

  // Q fragments (16 queries x 64 head-dim -> two K=32 A-frags)
  v16h aQ0 = load_frag_a(qhW, HD, q0 + l16, 0, h8);
  v16h aQ1 = load_frag_a(qhW, HD, q0 + l16, 32, h8);

  float preRow[8], postRow[8], effslope = 0.f;
#pragma unroll
  for (int j = 0; j < 8; ++j) {
    preRow[j]  = pre_th[w * 8 + j];
    postRow[j] = post_th[w * 8 + j];
    effslope  += preRow[j] * exp2f(-(float)(j + 1));  // pre-mixed ALiBi slope
  }

  float mrow[8], lrow[8];
  v8f O0 = {}, O1 = {}, O2 = {}, O3 = {};
#pragma unroll
  for (int r = 0; r < 8; ++r) { mrow[r] = -3.0e38f; lrow[r] = 0.f; }

  const float scale = 0.044194173824159216f;  // 1/sqrt(E=512)

  for (int c = 0; c < KVLEN / 32; ++c) {
    const int k0 = c * 32;
    if (k0 + 32 < KVLEN) {                    // prefetch next K chunk
      __builtin_prefetch(khW + (size_t)(k0 + 32) * HD, 0, 1);
      __builtin_prefetch(vTW + (k0 + 32), 0, 1);
    }
    // ---- energy for head w: two 16-key n-tiles, contraction over D=64 ----
    v8f e0 = {}, e1 = {};
    {
      v16h b = load_frag_bt(khW, HD, k0 + l16, 0, h16);
      e0 = wmma_f16(aQ0, b, e0);
      b = load_frag_bt(khW, HD, k0 + l16, 32, h16);
      e0 = wmma_f16(aQ1, b, e0);
      b = load_frag_bt(khW, HD, k0 + 16 + l16, 0, h16);
      e1 = wmma_f16(aQ0, b, e1);
      b = load_frag_bt(khW, HD, k0 + 16 + l16, 32, h16);
      e1 = wmma_f16(aQ1, b, e1);
    }
#pragma unroll
    for (int r = 0; r < 8; ++r) {             // publish raw energies
      ldsE[w][r + h8][l16]      = e0[r];
      ldsE[w][r + h8][16 + l16] = e1[r];
    }
    __syncthreads();

    // ---- pre-talking-heads mix + ALiBi + mask + online softmax ----
    float p0[8], p1[8], fac[8];
#pragma unroll
    for (int r = 0; r < 8; ++r) {
      const int q = q0 + r + h8;
      float v0 = 0.f, v1 = 0.f;
#pragma unroll
      for (int j = 0; j < 8; ++j) {
        v0 += preRow[j] * ldsE[j][r + h8][l16];
        v1 += preRow[j] * ldsE[j][r + h8][16 + l16];
      }
      const int kk0 = k0 + l16, kk1 = k0 + 16 + l16;
      if (kk0 < SQ) v0 -= effslope * fabsf((float)(q - kk0));
      if (kk1 < SQ) v1 -= effslope * fabsf((float)(q - kk1));
      bool keep0 = (kk0 < SQ) ? (maskN[(size_t)q * SQ + kk0] != 0) : (kk0 < SQ + 16);
      bool keep1 = (kk1 < SQ) ? (maskN[(size_t)q * SQ + kk1] != 0) : (kk1 < SQ + 16);
      if (!keep0) v0 = -1.0e4f;
      if (!keep1) v1 = -1.0e4f;
      v0 *= scale; v1 *= scale;
      // row max over 32 keys (16 lanes within a half hold the row)
      float t = fmaxf(v0, v1);
      t = fmaxf(t, __shfl_xor(t, 1, 32));
      t = fmaxf(t, __shfl_xor(t, 2, 32));
      t = fmaxf(t, __shfl_xor(t, 4, 32));
      t = fmaxf(t, __shfl_xor(t, 8, 32));
      const float mnew = fmaxf(mrow[r], t);
      const float f = __expf(mrow[r] - mnew);
      const float a0 = __expf(v0 - mnew);
      const float a1 = __expf(v1 - mnew);
      float s = a0 + a1;
      s += __shfl_xor(s, 1, 32);
      s += __shfl_xor(s, 2, 32);
      s += __shfl_xor(s, 4, 32);
      s += __shfl_xor(s, 8, 32);
      lrow[r] = lrow[r] * f + s;
      mrow[r] = mnew;
      fac[r]  = f;
      p0[r] = a0; p1[r] = a1;
    }
#pragma unroll
    for (int r = 0; r < 8; ++r) {             // rescale flash accumulators
      O0[r] *= fac[r]; O1[r] *= fac[r]; O2[r] *= fac[r]; O3[r] *= fac[r];
    }
#pragma unroll
    for (int r = 0; r < 8; ++r) {             // D-layout -> LDS (own region)
      ldsP[w][r + h8][l16]      = (_Float16)p0[r];
      ldsP[w][r + h8][16 + l16] = (_Float16)p1[r];
    }
    __syncthreads();                          // also guards ldsE reuse

    // ---- P·V : contraction over 32 keys, 4 d-tiles of 16 ----
    v16h aP = load_frag_a(&ldsP[w][0][0], 32, l16, 0, h8);
    v16h bv = load_frag_bt(vTW, KVLEN, l16, k0, h16);
    O0 = wmma_f16(aP, bv, O0);
    bv = load_frag_bt(vTW, KVLEN, 16 + l16, k0, h16);
    O1 = wmma_f16(aP, bv, O1);
    bv = load_frag_bt(vTW, KVLEN, 32 + l16, k0, h16);
    O2 = wmma_f16(aP, bv, O2);
    bv = load_frag_bt(vTW, KVLEN, 48 + l16, k0, h16);
    O3 = wmma_f16(aP, bv, O3);
  }

  // ---- normalize per source head, exchange, post-talking-heads mix ----
#pragma unroll
  for (int r = 0; r < 8; ++r) {
    const float inv = 1.0f / lrow[r];
    ldsO[w][r + h8][l16]      = O0[r] * inv;
    ldsO[w][r + h8][16 + l16] = O1[r] * inv;
    ldsO[w][r + h8][32 + l16] = O2[r] * inv;
    ldsO[w][r + h8][48 + l16] = O3[r] * inv;
  }
  __syncthreads();
  _Float16* xoW = xh + ((size_t)n * SQ + q0) * EMB + w * HD;
  for (int idx = lane; idx < 16 * 64; idx += 32) {
    const int row = idx >> 6, d = idx & 63;
    float acc = 0.f;
#pragma unroll
    for (int i = 0; i < 8; ++i) acc += postRow[i] * ldsO[i][row][d];
    xoW[(size_t)row * EMB + d] = (_Float16)acc;
  }
}

// ---------------------------------------------------------------------------
// Kernel 3: out = X @ Wo^T + bo  (16384 x 512 x 512), f16 WMMA, f32 accum.
// Each wave computes one 16x16 tile; 8 waves per block.
// ---------------------------------------------------------------------------
__global__ __launch_bounds__(256) void outproj_kernel(
    const _Float16* __restrict__ xh, const _Float16* __restrict__ WoH,
    const float* __restrict__ bo, float* __restrict__ out) {
  const int tid  = threadIdx.x;
  const int w    = tid >> 5;
  const int lane = tid & 31;
  const int hlf  = lane >> 4;
  const int l16  = lane & 15;
  const int h8 = hlf * 8, h16 = hlf * 16;

  const int tile = blockIdx.x * 8 + w;  // (16384/16) * (512/16) = 32768 tiles
  const int mt = tile >> 5;             // token tile 0..1023
  const int nt = tile & 31;             // eout tile  0..31
  const int row = mt * 16 + l16;
  const int col = nt * 16 + l16;

  v8f acc = {};
#pragma unroll 4
  for (int kc = 0; kc < 16; ++kc) {
    v16h a = load_frag_a(xh, EMB, row, kc * 32, h8);
    v16h b = load_frag_bt(WoH, EMB, col, kc * 32, h16);  // Wo row-major = B^T
    acc = wmma_f16(a, b, acc);
  }
  const float bias = bo[col];
#pragma unroll
  for (int r = 0; r < 8; ++r) {
    const int token = mt * 16 + r + h8;
    out[(size_t)token * EMB + col] = acc[r] + bias;
  }
}

// ---------------------------------------------------------------------------
extern "C" void kernel_launch(void* const* d_in, const int* in_sizes, int n_in,
                              void* d_out, int out_size, void* d_ws,
                              size_t ws_size, hipStream_t stream) {
  const float* vals   = (const float*)d_in[0];
  const float* keys   = (const float*)d_in[1];
  const float* qrs    = (const float*)d_in[2];
  const int*   mask   = (const int*)d_in[3];
  const float* Wv     = (const float*)d_in[4];
  const float* Wk     = (const float*)d_in[5];
  const float* Wq     = (const float*)d_in[6];
  const float* Wo     = (const float*)d_in[7];
  const float* bo     = (const float*)d_in[8];
  const float* pre_th = (const float*)d_in[9];
  const float* post_th= (const float*)d_in[10];
  const float* pk     = (const float*)d_in[11];
  const float* pv     = (const float*)d_in[12];
  float* out = (float*)d_out;

  char* ws = (char*)d_ws;
  size_t off = 0;
  auto take = [&](size_t bytes) {
    size_t o = off;
    off = (off + bytes + 255) & ~(size_t)255;
    return o;
  };
  _Float16* qh  = (_Float16*)(ws + take((size_t)8 * 8 * SQ * HD * 2));
  _Float16* kh  = (_Float16*)(ws + take((size_t)8 * 8 * KVLEN * HD * 2));
  _Float16* vT  = (_Float16*)(ws + take((size_t)8 * 8 * HD * KVLEN * 2));
  _Float16* xh  = (_Float16*)(ws + take((size_t)8 * SQ * EMB * 2));
  _Float16* WoH = (_Float16*)(ws + take((size_t)EMB * EMB * 2));
  (void)in_sizes; (void)n_in; (void)out_size; (void)ws_size;

  proj_kernel<<<8 * SQ, 256, 0, stream>>>(vals, keys, qrs, Wv, Wk, Wq,
                                          qh, kh, vT);
  persist_kernel<<<(8 * 8 * 32 * 64 + 255) / 256, 256, 0, stream>>>(pk, pv,
                                                                    kh, vT);
  cvt_wo_kernel<<<(EMB * EMB + 255) / 256, 256, 0, stream>>>(Wo, WoH);
  attn_kernel<<<8 * (SQ / 16), 256, 0, stream>>>(qh, kh, vT, mask, pre_th,
                                                 post_th, xh);
  outproj_kernel<<<(1024 * 32) / 8, 256, 0, stream>>>(xh, WoH, bo, out);
}
